// GRUDecoder_16329465659472
// MI455X (gfx1250) — matrix operand order repaired and checked by reading, not compiled, not run on hardware
//
#include <hip/hip_runtime.h>
#include <hip/hip_bf16.h>

typedef __bf16 bf16;
typedef __attribute__((ext_vector_type(16))) __bf16 v16bf;
typedef __attribute__((ext_vector_type(8)))  __bf16 v8bf;
typedef __attribute__((ext_vector_type(8)))  float  v8f;

#define BB   64
#define NN   49
#define ENC  2048
#define VV   32000
#define EE   512
#define AA   512
#define DD   512
#define TT   20
#define T1   21
#define NH   8
#define HD   64

// ---------------------------------------------------------------------------
// Pack fp32 weight into WMMA-B fragment-native bf16 layout.
// Logical matrix is K x N (K = reduction dim). trans=0: W stored (K,N)
// row-major.  trans=1: W stored (N,K) row-major (i.e. we use W^T).
// Fragment (kt,nt) of shape 32x16 is stored contiguously (512 elems):
//   idx-in-frag = lane*16 + e ; lanes 0-15: K=kt*32+e,     N=nt*16+lane
//                              lanes16-31: K=kt*32+16+e,   N=nt*16+lane-16
// ---------------------------------------------------------------------------
__global__ void pack_w(const float* __restrict__ W, bf16* __restrict__ P,
                       int K, int N, int trans) {
  size_t idx = (size_t)blockIdx.x * blockDim.x + threadIdx.x;
  size_t total = (size_t)K * N;
  if (idx >= total) return;
  int numKt = K >> 5;
  size_t frag = idx >> 9;
  int w = (int)(idx & 511);
  int lane = w >> 4, e = w & 15;
  int kt = (int)(frag % numKt);
  int nt = (int)(frag / numKt);
  int k = kt * 32 + ((lane < 16) ? 0 : 16) + e;
  int n = nt * 16 + (lane & 15);
  float v = trans ? W[(size_t)n * K + k] : W[(size_t)k * N + n];
  P[idx] = (bf16)v;
}

// ---------------------------------------------------------------------------
// Generic bf16 WMMA GEMM: out = A(MxK, bf16 row-major) * packedB(KxN) + bias
// 8 waves per block; each wave computes a 16x64 tile (4 accumulators), so one
// A-fragment load feeds 4 v_wmma ops. Inner loop is software-pipelined: the
// next K-tile's A + 4xB fragments are loaded into a second register set
// before the current tile's WMMAs issue, so matrix ops overlap the loads.
// B streams additionally prefetched 2 K-tiles ahead (global_prefetch_b8).
// Requires: M % 16 == 0, N % 64 == 0, K % 32 == 0 (and K >= 64).
// ---------------------------------------------------------------------------
__global__ void gemm_bf16(const bf16* __restrict__ A, int lda,
                          const bf16* __restrict__ Bp,
                          const float* __restrict__ bias,
                          float* outF, long long ldF,
                          bf16* outH, long long ldH,
                          int M, int N, int K) {
  int lane = threadIdx.x & 31;
  int wave = threadIdx.x >> 5;
  int ntBase = (blockIdx.x * 8 + wave) * 4;  // first of four 16-col tiles
  int mt = blockIdx.y;
  int m0 = mt << 4, n0 = ntBase << 4;
  if (n0 >= N) return;
  int numKt = K >> 5;

  int row  = m0 + (lane & 15);
  int koff = (lane < 16) ? 0 : 16;
  const bf16* aptr = A + (size_t)row * lda + koff;
  size_t tileStride = (size_t)numKt * 512;
  const bf16* bp0 = Bp + (size_t)(ntBase + 0) * tileStride + lane * 16;
  const bf16* bp1 = Bp + (size_t)(ntBase + 1) * tileStride + lane * 16;
  const bf16* bp2 = Bp + (size_t)(ntBase + 2) * tileStride + lane * 16;
  const bf16* bp3 = Bp + (size_t)(ntBase + 3) * tileStride + lane * 16;

  union Afrag { v16bf v; v8bf h[2]; };

  v8f c0 = {}, c1 = {}, c2 = {}, c3 = {};

  // prologue: load K-tile 0
  Afrag a;
  a.h[0] = *(const v8bf*)(aptr);
  a.h[1] = *(const v8bf*)(aptr + 8);
  v16bf b0 = *(const v16bf*)(bp0);
  v16bf b1 = *(const v16bf*)(bp1);
  v16bf b2 = *(const v16bf*)(bp2);
  v16bf b3 = *(const v16bf*)(bp3);

  for (int kt = 0; kt < numKt - 1; ++kt) {
    aptr += 32;
    bp0 += 512; bp1 += 512; bp2 += 512; bp3 += 512;
    // issue next tile's loads into fresh registers
    Afrag an;
    an.h[0] = *(const v8bf*)(aptr);
    an.h[1] = *(const v8bf*)(aptr + 8);
    v16bf d0 = *(const v16bf*)(bp0);
    v16bf d1 = *(const v16bf*)(bp1);
    v16bf d2 = *(const v16bf*)(bp2);
    v16bf d3 = *(const v16bf*)(bp3);
    __builtin_prefetch(bp0 + 1024, 0, 1);   // 2 K-tiles ahead
    __builtin_prefetch(bp2 + 1024, 0, 1);
    // current tile's matrix ops overlap the outstanding loads
    c0 = __builtin_amdgcn_wmma_f32_16x16x32_bf16(false, a.v, false, b0,
                                                 (short)0, c0, false, false);
    c1 = __builtin_amdgcn_wmma_f32_16x16x32_bf16(false, a.v, false, b1,
                                                 (short)0, c1, false, false);
    c2 = __builtin_amdgcn_wmma_f32_16x16x32_bf16(false, a.v, false, b2,
                                                 (short)0, c2, false, false);
    c3 = __builtin_amdgcn_wmma_f32_16x16x32_bf16(false, a.v, false, b3,
                                                 (short)0, c3, false, false);
    a = an; b0 = d0; b1 = d1; b2 = d2; b3 = d3;
  }
  // epilogue: last K-tile
  c0 = __builtin_amdgcn_wmma_f32_16x16x32_bf16(false, a.v, false, b0,
                                               (short)0, c0, false, false);
  c1 = __builtin_amdgcn_wmma_f32_16x16x32_bf16(false, a.v, false, b1,
                                               (short)0, c1, false, false);
  c2 = __builtin_amdgcn_wmma_f32_16x16x32_bf16(false, a.v, false, b2,
                                               (short)0, c2, false, false);
  c3 = __builtin_amdgcn_wmma_f32_16x16x32_bf16(false, a.v, false, b3,
                                               (short)0, c3, false, false);

  int rbase = m0 + ((lane < 16) ? 0 : 8);
  v8f acc[4] = { c0, c1, c2, c3 };
#pragma unroll
  for (int tile = 0; tile < 4; ++tile) {
    int col = n0 + tile * 16 + (lane & 15);
    float bv = bias ? bias[col] : 0.0f;
#pragma unroll
    for (int r = 0; r < 8; ++r) {
      float val = acc[tile][r] + bv;
      size_t rr = (size_t)(rbase + r);
      if (outF) outF[rr * (size_t)ldF + col] = val;
      if (outH) outH[rr * (size_t)ldH + col] = (bf16)val;
    }
  }
}

// mean over N of encoder outputs -> bf16 (B x ENC)
__global__ void enc_mean(const float* __restrict__ enc, bf16* __restrict__ mbf) {
  int idx = blockIdx.x * blockDim.x + threadIdx.x;
  if (idx >= BB * ENC) return;
  int b = idx / ENC, cc = idx % ENC;
  const float* p = enc + (size_t)b * NN * ENC + cc;
  float s = 0.f;
  for (int n = 0; n < NN; ++n) s += p[(size_t)n * ENC];
  mbf[idx] = (bf16)(s * (1.0f / NN));
}

__global__ void f32_to_bf16(const float* __restrict__ s, bf16* __restrict__ d, size_t n) {
  size_t i = (size_t)blockIdx.x * blockDim.x + threadIdx.x;
  if (i < n) d[i] = (bf16)s[i];
}

// embedding gather for step t into x[:, 0:E]
__global__ void embed_step(const float* __restrict__ embW, const int* __restrict__ cap,
                           bf16* __restrict__ xbf, int t) {
  int idx = blockIdx.x * blockDim.x + threadIdx.x;
  if (idx >= BB * EE) return;
  int b = idx / EE, e = idx % EE;
  int tok = cap[b * T1 + t];
  xbf[(size_t)b * (EE + ENC) + e] = (bf16)embW[(size_t)tok * EE + e];
}

// attention: one block per (b,h), 64 threads
__global__ void attention(const float* __restrict__ q, const float* __restrict__ kb,
                          const float* __restrict__ vb, float* __restrict__ alpha,
                          bf16* __restrict__ ctxbf) {
  int bh = blockIdx.x;
  int b = bh >> 3, h = bh & 7;
  int d = threadIdx.x;
  __shared__ float qs[HD];
  __shared__ float sc[64];
  __shared__ float sm;
  qs[d] = q[(size_t)b * AA + h * HD + d];
  __syncthreads();
  if (d < NN) {
    const float* kp = kb + ((size_t)(b * NN + d)) * AA + h * HD;
    float s = 0.f;
    for (int i = 0; i < HD; ++i) s += qs[i] * kp[i];
    sc[d] = s * 0.125f;   // 1/sqrt(64)
  }
  __syncthreads();
  if (d == 0) {
    float m = -1e30f;
    for (int n = 0; n < NN; ++n) m = fmaxf(m, sc[n]);
    float s = 0.f;
    for (int n = 0; n < NN; ++n) { float e = __expf(sc[n] - m); sc[n] = e; s += e; }
    sm = s;
  }
  __syncthreads();
  float inv = 1.0f / sm;
  if (d < NN) alpha[(size_t)bh * NN + d] = sc[d] * inv;
  float acc = 0.f;
  for (int n = 0; n < NN; ++n)
    acc += sc[n] * inv * vb[((size_t)(b * NN + n)) * AA + h * HD + d];
  ctxbf[(size_t)b * AA + h * HD + d] = (bf16)acc;
}

// mean over heads of alpha -> attention_weights[b, t, n]
__global__ void attn_mean(const float* __restrict__ alpha, float* __restrict__ outA, int t) {
  int idx = blockIdx.x * blockDim.x + threadIdx.x;
  if (idx >= BB * NN) return;
  int b = idx / NN, n = idx % NN;
  float s = 0.f;
  for (int h = 0; h < NH; ++h) s += alpha[((size_t)(b * NH + h)) * NN + n];
  outA[((size_t)b * TT + t) * NN + n] = s * 0.125f;
}

// fused GRU gates + LayerNorm; h updated in place, bf16 copy refreshed
__global__ void gru_ln(const float* __restrict__ gi, const float* __restrict__ gh,
                       const float* __restrict__ lng, const float* __restrict__ lnb,
                       float* __restrict__ hbuf, bf16* __restrict__ hbf) {
  int b = blockIdx.x;
  int d = threadIdx.x;   // 512 threads
  size_t o = (size_t)b * 3 * DD;
  float ir = gi[o + d], iz = gi[o + DD + d], in_ = gi[o + 2 * DD + d];
  float hr = gh[o + d], hz = gh[o + DD + d], hn  = gh[o + 2 * DD + d];
  float r = 1.f / (1.f + __expf(-(ir + hr)));
  float z = 1.f / (1.f + __expf(-(iz + hz)));
  float n = tanhf(in_ + r * hn);
  float hp = hbuf[(size_t)b * DD + d];
  float hnew = (1.f - z) * n + z * hp;

  __shared__ float red[DD];
  red[d] = hnew; __syncthreads();
  for (int s = DD / 2; s > 0; s >>= 1) { if (d < s) red[d] += red[d + s]; __syncthreads(); }
  float mu = red[0] * (1.0f / DD);
  __syncthreads();
  float df = hnew - mu;
  red[d] = df * df; __syncthreads();
  for (int s = DD / 2; s > 0; s >>= 1) { if (d < s) red[d] += red[d + s]; __syncthreads(); }
  float var = red[0] * (1.0f / DD);
  float hl = df * rsqrtf(var + 1e-5f) * lng[d] + lnb[d];
  hbuf[(size_t)b * DD + d] = hl;
  hbf [(size_t)b * DD + d] = (bf16)hl;
}

// ---------------------------------------------------------------------------
static inline void launch_gemm(hipStream_t s, const bf16* A, int lda, const bf16* Bp,
                               const float* bias, float* outF, long long ldF,
                               bf16* outH, long long ldH, int M, int N, int K) {
  dim3 g((unsigned)((N + 511) / 512), (unsigned)(M / 16));
  gemm_bf16<<<g, 256, 0, s>>>(A, lda, Bp, bias, outF, ldF, outH, ldH, M, N, K);
}

static inline void launch_pack(hipStream_t s, const float* W, bf16* P, int K, int N, int tr) {
  size_t tot = (size_t)K * N;
  pack_w<<<(unsigned)((tot + 255) / 256), 256, 0, s>>>(W, P, K, N, tr);
}

extern "C" void kernel_launch(void* const* d_in, const int* in_sizes, int n_in,
                              void* d_out, int out_size, void* d_ws, size_t ws_size,
                              hipStream_t stream) {
  (void)in_sizes; (void)n_in; (void)out_size; (void)ws_size;
  const float* enc   = (const float*)d_in[0];
  const int*   cap   = (const int*)  d_in[1];
  const float* embW  = (const float*)d_in[2];
  const float* Wq    = (const float*)d_in[3];
  const float* bq    = (const float*)d_in[4];
  const float* Wk    = (const float*)d_in[5];
  const float* bk    = (const float*)d_in[6];
  const float* Wv    = (const float*)d_in[7];
  const float* bv    = (const float*)d_in[8];
  const float* Wo    = (const float*)d_in[9];
  const float* bo    = (const float*)d_in[10];
  const float* initw = (const float*)d_in[11];
  const float* initb = (const float*)d_in[12];
  const float* Wih   = (const float*)d_in[13];
  const float* bih   = (const float*)d_in[14];
  const float* Whh   = (const float*)d_in[15];
  const float* bhh   = (const float*)d_in[16];
  const float* lng   = (const float*)d_in[17];
  const float* lnb   = (const float*)d_in[18];
  const float* fcw   = (const float*)d_in[19];
  const float* fcb   = (const float*)d_in[20];

  float* outLogits = (float*)d_out;                               // (B,T,V)
  float* outAttn   = outLogits + (size_t)BB * TT * VV;            // (B,T,N)

  // ---- workspace carve-up (256B aligned) ----
  char* w = (char*)d_ws;
  size_t off = 0;
  auto alloc = [&](size_t bytes) -> void* {
    void* p = w + off;
    off = (off + bytes + 255) & ~(size_t)255;
    return p;
  };
  bf16* pWq   = (bf16*)alloc((size_t)DD  * AA  * 2);
  bf16* pWk   = (bf16*)alloc((size_t)ENC * AA  * 2);
  bf16* pWv   = (bf16*)alloc((size_t)ENC * AA  * 2);
  bf16* pWo   = (bf16*)alloc((size_t)AA  * ENC * 2);
  bf16* pInit = (bf16*)alloc((size_t)ENC * DD  * 2);
  bf16* pWih  = (bf16*)alloc((size_t)(EE + ENC) * 3 * DD * 2);
  bf16* pWhh  = (bf16*)alloc((size_t)DD  * 3 * DD * 2);
  bf16* pFc   = (bf16*)alloc((size_t)DD  * VV * 2);
  bf16* encbf = (bf16*)alloc((size_t)BB * NN * ENC * 2);
  bf16* meanbf= (bf16*)alloc((size_t)BB * ENC * 2);
  float* kbuf = (float*)alloc((size_t)BB * NN * AA * 4);
  float* vbuf = (float*)alloc((size_t)BB * NN * AA * 4);
  float* hbuf = (float*)alloc((size_t)BB * DD * 4);
  bf16*  hbf  = (bf16*) alloc((size_t)BB * DD * 2);
  float* qbuf = (float*)alloc((size_t)BB * AA * 4);
  float* alpha= (float*)alloc((size_t)BB * NH * NN * 4);
  bf16*  ctxbf= (bf16*) alloc((size_t)BB * AA * 2);
  bf16*  xbf  = (bf16*) alloc((size_t)BB * (EE + ENC) * 2);
  float* gi   = (float*)alloc((size_t)BB * 3 * DD * 4);
  float* gh   = (float*)alloc((size_t)BB * 3 * DD * 4);

  // ---- one-time prep ----
  launch_pack(stream, Wq,    pWq,   DD,       AA,     0);
  launch_pack(stream, Wk,    pWk,   ENC,      AA,     0);
  launch_pack(stream, Wv,    pWv,   ENC,      AA,     0);
  launch_pack(stream, Wo,    pWo,   AA,       ENC,    0);
  launch_pack(stream, initw, pInit, ENC,      DD,     1);
  launch_pack(stream, Wih,   pWih,  EE + ENC, 3 * DD, 1);
  launch_pack(stream, Whh,   pWhh,  DD,       3 * DD, 1);
  launch_pack(stream, fcw,   pFc,   DD,       VV,     1);

  {
    size_t tot = (size_t)BB * NN * ENC;
    f32_to_bf16<<<(unsigned)((tot + 255) / 256), 256, 0, stream>>>(enc, encbf, tot);
  }
  enc_mean<<<(BB * ENC + 255) / 256, 256, 0, stream>>>(enc, meanbf);

  // h0 = mean @ init_w.T + init_b
  launch_gemm(stream, meanbf, ENC, pInit, initb, hbuf, DD, hbf, DD, BB, DD, ENC);
  // K, V projections over all (b,n)
  launch_gemm(stream, encbf, ENC, pWk, bk, kbuf, AA, nullptr, 0, BB * NN, AA, ENC);
  launch_gemm(stream, encbf, ENC, pWv, bv, vbuf, AA, nullptr, 0, BB * NN, AA, ENC);

  // ---- decode loop ----
  for (int t = 0; t < TT; ++t) {
    // q = h @ Wq + bq
    launch_gemm(stream, hbf, DD, pWq, bq, qbuf, AA, nullptr, 0, BB, AA, DD);
    // attention per (b,h)
    attention<<<BB * NH, HD, 0, stream>>>(qbuf, kbuf, vbuf, alpha, ctxbf);
    attn_mean<<<(BB * NN + 255) / 256, 256, 0, stream>>>(alpha, outAttn, t);
    // x = [emb_t , ctx @ Wo + bo]
    embed_step<<<(BB * EE + 255) / 256, 256, 0, stream>>>(embW, cap, xbf, t);
    launch_gemm(stream, ctxbf, AA, pWo, bo, nullptr, 0, xbf + EE, EE + ENC, BB, ENC, AA);
    // GRU gate pre-activations
    launch_gemm(stream, xbf, EE + ENC, pWih, bih, gi, 3 * DD, nullptr, 0, BB, 3 * DD, EE + ENC);
    launch_gemm(stream, hbf, DD,       pWhh, bhh, gh, 3 * DD, nullptr, 0, BB, 3 * DD, DD);
    // gates + LN, h updated in place
    gru_ln<<<BB, DD, 0, stream>>>(gi, gh, lng, lnb, hbuf, hbf);
    // logits directly into output slice [:, t, :]
    launch_gemm(stream, hbf, DD, pFc, fcb, outLogits + (size_t)t * VV,
                (long long)TT * VV, nullptr, 0, BB, VV, DD);
  }
}
